// ChannelEmbedding_1159641170097
// MI455X (gfx1250) — matrix operand order repaired and checked by reading, not compiled, not run on hardware
//
#include <hip/hip_runtime.h>

typedef _Float16 v16h __attribute__((ext_vector_type(16)));
typedef _Float16 v8h  __attribute__((ext_vector_type(8)));
typedef float    v8f  __attribute__((ext_vector_type(8)));

#define Bn   64
#define Cn   9
#define Tn   8192
#define PCDn 16
#define En   128
#define Kn   7
#define CP   144          // C * PCD
#define LSTR 168          // padded LDS row stride (f16 elems): 160 (K pad) + 8 (bank pad)
#define TT   128          // t-tile per workgroup
#define NTHREADS 256

__global__ __launch_bounds__(NTHREADS) void ce_fused_kernel(
    const float* __restrict__ x,
    const float* __restrict__ w_conv,
    const float* __restrict__ b_conv,
    const float* __restrict__ w_proj,
    const float* __restrict__ b_proj,
    const float* __restrict__ gamma,
    const float* __restrict__ beta,
    float* __restrict__ out)
{
    __shared__ _Float16 sY[TT * LSTR];     // conv output tile, f16, [t][k] k-padded w/ zeros
    __shared__ _Float16 sW[En * LSTR];     // w_proj as f16, [e][k] k-padded w/ zeros
    __shared__ float    sX[Cn * 144];      // x tile + halo(3,3), zero padded at seq edges
    __shared__ float    sWc[CP * 8];       // [cp][0..6]=w_conv taps, [7]=b_conv

    const int tid = threadIdx.x;
    const int t0  = blockIdx.x * TT;
    const int b   = blockIdx.y;

    // ---------------- Phase 1: stage inputs into LDS ----------------
    for (int i = tid; i < Cn * 144; i += NTHREADS) {
        int c  = i / 144;
        int tl = i % 144;                  // == t_local + 3
        int t  = t0 + tl - 3;
        float v = 0.f;
        if (t >= 0 && t < Tn) v = x[(b * Cn + c) * Tn + t];
        sX[i] = v;
    }
    for (int i = tid; i < CP * 8; i += NTHREADS) {
        int cp = i >> 3, k = i & 7;
        sWc[i] = (k < Kn) ? w_conv[cp * Kn + k] : b_conv[cp];
    }
    for (int i = tid; i < En * LSTR; i += NTHREADS) {
        int e = i / LSTR, k = i % LSTR;
        sW[i] = (_Float16)((k < CP) ? w_proj[e * CP + k] : 0.f);
    }
    for (int i = tid; i < TT * (LSTR - CP); i += NTHREADS) {
        int t = i / (LSTR - CP), k = CP + i % (LSTR - CP);
        sY[t * LSTR + k] = (_Float16)0.f;
    }
    __syncthreads();

    // ------- Phase 2: grouped conv (1->16 per ch, k=7, pad=3) + bias + ReLU -> f16 -------
    if (tid < CP) {
        const int cp = tid;
        const int c  = cp >> 4;            // PCD == 16
        const float* xr = &sX[c * 144];
        const float* wc = &sWc[cp * 8];
        const float w0=wc[0], w1=wc[1], w2=wc[2], w3=wc[3], w4=wc[4], w5=wc[5], w6=wc[6], bias=wc[7];
        float x0=xr[0], x1=xr[1], x2=xr[2], x3=xr[3], x4=xr[4], x5=xr[5], x6=xr[6];
        #pragma unroll 4
        for (int tl = 0; tl < TT; ++tl) {
            float a = bias;
            a = fmaf(w0, x0, a); a = fmaf(w1, x1, a); a = fmaf(w2, x2, a);
            a = fmaf(w3, x3, a); a = fmaf(w4, x4, a); a = fmaf(w5, x5, a);
            a = fmaf(w6, x6, a);
            sY[tl * LSTR + cp] = (_Float16)fmaxf(a, 0.f);
            x0=x1; x1=x2; x2=x3; x3=x4; x4=x5; x5=x6; x6 = xr[tl + 7];
        }
    }
    __syncthreads();

    // ------- Phase 3: per-wave WMMA GEMM  Z(16t x 128e) = Y(16x160) x W^T(160x128) -------
    const int wave = tid >> 5;
    const int lane = tid & 31;
    const int hrow = lane & 15;            // A: t-row; B: e-col within tile
    const int half = lane >> 4;            // lane group selects K sub-range

    union AV { v16h v; v8h h[2]; };

    v8f acc[8] = {};
    const _Float16* ybase = &sY[(wave * 16 + hrow) * LSTR];

    for (int kc = 0; kc < 5; ++kc) {       // K = 5 * 32 = 160 (zero-padded 144)
        AV a;                              // A 16x32 f16: lanes0-15 K 0..15 ; lanes16-31 K 16..31
        a.h[0] = *(const v8h*)(ybase + kc * 32 + half * 16);
        a.h[1] = *(const v8h*)(ybase + kc * 32 + half * 16 + 8);
        #pragma unroll
        for (int j = 0; j < 8; ++j) {      // B 32x16 f16: lanes0-15 K 0..15 ; lanes16-31 K 16..31
            const _Float16* wb = &sW[(j * 16 + hrow) * LSTR + kc * 32 + half * 16];
            AV bm;
            bm.h[0] = *(const v8h*)(wb);
            bm.h[1] = *(const v8h*)(wb + 8);
            acc[j] = __builtin_amdgcn_wmma_f32_16x16x32_f16(
                false, a.v, false, bm.v, (short)0, acc[j], false, false);
        }
    }

    // ------- Phase 4: + b_proj, LayerNorm over E=128, affine, coalesced store -------
    float bp[8], gm[8], bt[8];
    #pragma unroll
    for (int j = 0; j < 8; ++j) {
        const int e = j * 16 + hrow;
        bp[j] = b_proj[e];
        gm[j] = gamma[e];
        bt[j] = beta[e];
    }
    #pragma unroll
    for (int j = 0; j < 8; ++j) {
        #pragma unroll
        for (int v = 0; v < 8; ++v) acc[j][v] += bp[j];
    }

    const int trow_base = t0 + wave * 16 + half * 8;   // C/D layout: lanes0-15 -> M=v, lanes16-31 -> M=v+8
    #pragma unroll
    for (int v = 0; v < 8; ++v) {
        float s = 0.f, s2 = 0.f;
        #pragma unroll
        for (int j = 0; j < 8; ++j) {
            const float z = acc[j][v];
            s += z; s2 = fmaf(z, z, s2);
        }
        #pragma unroll
        for (int off = 1; off < 16; off <<= 1) {       // reduce within each 16-lane half
            s  += __shfl_xor(s,  off, 32);
            s2 += __shfl_xor(s2, off, 32);
        }
        const float mean = s * (1.0f / En);
        const float var  = fmaf(s2, 1.0f / En, -mean * mean);
        const float inv  = rsqrtf(var + 1e-5f);
        float* orow = out + ((size_t)b * Tn + (size_t)(trow_base + v)) * En;
        #pragma unroll
        for (int j = 0; j < 8; ++j) {
            orow[j * 16 + hrow] = fmaf((acc[j][v] - mean) * inv, gm[j], bt[j]);
        }
    }
}

extern "C" void kernel_launch(void* const* d_in, const int* in_sizes, int n_in,
                              void* d_out, int out_size, void* d_ws, size_t ws_size,
                              hipStream_t stream) {
    (void)in_sizes; (void)n_in; (void)d_ws; (void)ws_size; (void)out_size;
    const float* x      = (const float*)d_in[0];
    const float* w_conv = (const float*)d_in[1];
    const float* b_conv = (const float*)d_in[2];
    const float* w_proj = (const float*)d_in[3];
    const float* b_proj = (const float*)d_in[4];
    const float* gamma  = (const float*)d_in[5];
    const float* beta   = (const float*)d_in[6];
    float* out = (float*)d_out;

    dim3 grid(Tn / TT, Bn);   // 64 x 64 workgroups
    ce_fused_kernel<<<grid, NTHREADS, 0, stream>>>(x, w_conv, b_conv, w_proj,
                                                   b_proj, gamma, beta, out);
}
